// GATSpatioTemporalModel_1005022347770
// MI455X (gfx1250) — matrix operand order repaired and checked by reading, not compiled, not run on hardware
//
#include <hip/hip_runtime.h>
#include <hip/hip_bf16.h>

typedef __attribute__((ext_vector_type(16))) __bf16 v16bf;
typedef __attribute__((ext_vector_type(8)))  float  v8f;

#define B_   8
#define N_   512
#define DIN_ 64
#define H_   8
#define F_   128
#define L_   2
#define NEGV  (-9e15f)
#define ALPHA 0.2f

// pack two fp32 -> two bf16 in one dword (lowered to v_cvt_pk_bf16_f32)
__device__ __forceinline__ unsigned pack2bf(float a, float b) {
  union { __bf16 h[2]; unsigned u; } r;
  r.h[0] = (__bf16)a;
  r.h[1] = (__bf16)b;
  return r.u;
}
__device__ __forceinline__ unsigned short f2bf(float f) {
  union { __bf16 h; unsigned short u; } r;
  r.h = (__bf16)f;
  return r.u;
}

union FragU { uint4 u[2]; v16bf v; };

// A fragment: 16x32 bf16, row-major tile in LDS (stride 32).
__device__ __forceinline__ v16bf load_frag_A16x32(const unsigned short* a, int lane) {
  int m  = lane & 15;
  int kh = (lane >> 4) * 8;
  FragU r;
  r.u[0] = *(const uint4*)(a + m * 32 + kh);
  r.u[1] = *(const uint4*)(a + m * 32 + 16 + kh);
  return r.v;
}

// B fragment: 32x16 bf16 from N-major staged tile Bt[n][k] (stride 32).
__device__ __forceinline__ v16bf load_frag_B32x16(const unsigned short* bt, int lane, int ncol0) {
  int n  = ncol0 + (lane & 15);
  int kb = (lane >> 4) * 8;
  FragU r;
  r.u[0] = *(const uint4*)(bt + n * 32 + kb);
  r.u[1] = *(const uint4*)(bt + n * 32 + 16 + kb);
  return r.v;
}

// ---------------------------------------------------------------------------
// Generic GEMM: C[M,128] = A[M,K] * B[K,128], bf16 WMMA, fp32 accumulate.
// grid.x = M/64, grid.y = groups. 256 threads = 8 waves.
// ABF16 path stages A via GLOBAL_LOAD_ASYNC_TO_LDS_B128 (ASYNCcnt-tracked).
// ---------------------------------------------------------------------------
template<int ABF16, int EPI>
__global__ void __launch_bounds__(256)
gemm128(const void* __restrict__ Aall, const float* __restrict__ Ball,
        float* __restrict__ Call, const float* __restrict__ bias,
        int M, int K, int H, int aGrp, int bGrp, int cGrp)
{
  __shared__ __align__(16) unsigned short Al[64 * 32];
  __shared__ __align__(16) unsigned short Bt[128 * 32];

  int g    = blockIdx.y;
  int row0 = blockIdx.x * 64;
  int aoff = (g / H) * aGrp;
  int boff = (g % H) * bGrp;
  int coff = g * cGrp;
  const float* Bg = Ball + boff;

  int t = threadIdx.x;
  int lane = t & 31, w = t >> 5;
  int rtile = w >> 1;          // 4 row subtiles of 16
  int c0    = (w & 1) * 4;     // 4 col subtiles of 16 per wave

  // staging coordinates (fixed per thread)
  int ar = t >> 2, ak0 = (t & 3) * 8;         // A: row 0..63, k chunk of 8
  int bj0 = (t >> 4) * 2, bn0 = (t & 15) * 8; // B: k-pair, 8 cols

  v8f acc[4] = {};

  for (int kc = 0; kc < K; kc += 32) {
    if constexpr (ABF16) {
      // async direct global->LDS copy of the bf16 A chunk (one b128 per lane)
      unsigned lds = (unsigned)(size_t)(Al + (ar * 32 + ak0));
      const unsigned short* gp =
          (const unsigned short*)Aall + aoff + (row0 + ar) * K + kc + ak0;
      asm volatile("global_load_async_to_lds_b128 %0, %1, off"
                   :: "v"(lds), "v"((unsigned long long)(size_t)gp)
                   : "memory");
    } else {
      // stage A chunk [64][32] -> bf16, one b128 LDS store per thread
      const float4* A = (const float4*)((const float*)Aall + aoff + (row0 + ar) * K + kc + ak0);
      float4 a0 = A[0], a1 = A[1];
      uint4 st;
      st.x = pack2bf(a0.x, a0.y); st.y = pack2bf(a0.z, a0.w);
      st.z = pack2bf(a1.x, a1.y); st.w = pack2bf(a1.z, a1.w);
      if (kc + 32 < K) __builtin_prefetch((const float*)A + 32, 0, 3);
      *(uint4*)(Al + (ar * 32 + ak0)) = st;
    }
    { // stage B transposed: Bt[n][j] = B[kc+j][n]; pack adjacent k rows -> b32
      const float4* r0 = (const float4*)(Bg + (kc + bj0) * 128 + bn0);
      const float4* r1 = (const float4*)(Bg + (kc + bj0 + 1) * 128 + bn0);
      float4 x0 = r0[0], x1 = r0[1], y0 = r1[0], y1 = r1[1];
      unsigned* dst = (unsigned*)(Bt + bn0 * 32 + bj0);
      dst[0]   = pack2bf(x0.x, y0.x);
      dst[16]  = pack2bf(x0.y, y0.y);
      dst[32]  = pack2bf(x0.z, y0.z);
      dst[48]  = pack2bf(x0.w, y0.w);
      dst[64]  = pack2bf(x1.x, y1.x);
      dst[80]  = pack2bf(x1.y, y1.y);
      dst[96]  = pack2bf(x1.z, y1.z);
      dst[112] = pack2bf(x1.w, y1.w);
      if (kc + 32 < K) __builtin_prefetch((const float*)r0 + 32 * 128, 0, 3);
    }
    if constexpr (ABF16) {
      asm volatile("s_wait_asynccnt 0" ::: "memory");
    }
    __syncthreads();
    v16bf af = load_frag_A16x32(Al + rtile * 16 * 32, lane);
    for (int i = 0; i < 4; ++i) {
      v16bf bfv = load_frag_B32x16(Bt, lane, (c0 + i) * 16);
      acc[i] = __builtin_amdgcn_wmma_f32_16x16x32_bf16(false, af, false, bfv,
                                                       (short)0, acc[i], false, false);
    }
    __syncthreads();
  }

  for (int i = 0; i < 4; ++i) {
    int col = (c0 + i) * 16 + (lane & 15);
    for (int rr = 0; rr < 8; ++rr) {
      int grow = row0 + rtile * 16 + rr + ((lane >> 4) << 3);
      float v = acc[i][rr];
      if (EPI == 1) { v += bias[col]; v = v > 0.f ? v : 0.f; }
      Call[coff + grow * 128 + col] = v;
    }
  }
}

// ---------------------------------------------------------------------------
// Attention scores: s1[g,n] = X[g,n,:].a[:F], s2 = X.a[F:]. One wave per row.
// ---------------------------------------------------------------------------
__global__ void __launch_bounds__(256)
score_kernel(const float* __restrict__ X, const float* __restrict__ avec,
             float* __restrict__ s1, float* __restrict__ s2, int H, int total)
{
  int t = threadIdx.x;
  int wid = (int)(blockIdx.x * blockDim.x + t) >> 5;
  if (wid >= total) return;
  int lane = t & 31;
  int g = wid / N_;
  const float* av = avec + (g % H) * (2 * F_);
  const float* x  = X + wid * F_;
  float a1 = 0.f, a2 = 0.f;
  for (int k = 0; k < 4; ++k) {
    int f = lane + 32 * k;
    float xv = x[f];
    a1 += xv * av[f];
    a2 += xv * av[F_ + f];
  }
  for (int d = 16; d > 0; d >>= 1) {
    a1 += __shfl_xor(a1, d, 32);
    a2 += __shfl_xor(a2, d, 32);
  }
  if (lane == 0) { s1[wid] = a1; s2[wid] = a2; }
}

// ---------------------------------------------------------------------------
// Fused masked-softmax attention * V (flash-style two-pass, [N,N] never
// materialized). MODE 0: ELU + bf16 concat into multi. MODE 1: fp32 out.
// ---------------------------------------------------------------------------
template<int MODE>
__global__ void __launch_bounds__(128)
attn_kernel(const float* __restrict__ s1, const float* __restrict__ s2,
            const int* __restrict__ adj, const float* __restrict__ V,
            void* __restrict__ Out, int H)
{
  __shared__ __align__(16) unsigned short At[16 * 32];
  __shared__ __align__(16) unsigned short Bt[128 * 32];
  __shared__ float redm[128], reds[128];
  __shared__ float rmax[16], rinv[16], s1sh[16];

  int g = blockIdx.y;
  int b = g / H, head = g % H;
  int n0 = blockIdx.x * 16;
  int t = threadIdx.x;
  int lane = t & 31, w = t >> 5;

  { // pass 1: per-row max & softmax denom (8 threads per row, online)
    int r = t >> 3, sub = t & 7;
    int n = n0 + r;
    float s1v = s1[g * N_ + n];
    if (sub == 0) s1sh[r] = s1v;
    const int* arow = adj + (b * N_ + n) * N_;
    const float* s2g = s2 + g * N_;
    float m = -3.0e38f, s = 0.f;
    for (int mm = sub; mm < N_; mm += 8) {
      float e = s1v + s2g[mm];
      e = e > 0.f ? e : ALPHA * e;
      float v = (arow[mm] > 0) ? e : NEGV;
      if (v > m) { s = s * __expf(m - v) + 1.f; m = v; }
      else        s += __expf(v - m);
    }
    redm[t] = m; reds[t] = s;
    __syncthreads();
    if (sub == 0) {
      for (int j = 1; j < 8; ++j) {
        float mj = redm[t + j], sj = reds[t + j];
        if (mj > m) { s = s * __expf(m - mj) + sj; m = mj; }
        else          s += sj * __expf(mj - m);
      }
      rmax[r] = m;
      rinv[r] = 1.f / s;
    }
    __syncthreads();
  }

  v8f acc[2] = {};
  const float* s2g = s2 + g * N_;
  int pr = t >> 3, pj0 = (t & 7) * 4;          // prob tile coords
  int vj0 = (t & 15) * 2, vf0 = (t >> 4) * 16; // V staging coords

  for (int mc = 0; mc < N_; mc += 32) {
    { // stage normalized probability tile [16][32] bf16 (uint2 store)
      float p[4];
      const int* arow = adj + (b * N_ + n0 + pr) * N_ + mc + pj0;
      float mx = rmax[pr], iv = rinv[pr], s1v = s1sh[pr];
      for (int e = 0; e < 4; ++e) {
        float ev = s1v + s2g[mc + pj0 + e];
        ev = ev > 0.f ? ev : ALPHA * ev;
        float vv = (arow[e] > 0) ? ev : NEGV;
        p[e] = __expf(vv - mx) * iv;
      }
      uint2 st; st.x = pack2bf(p[0], p[1]); st.y = pack2bf(p[2], p[3]);
      *(uint2*)(At + pr * 32 + pj0) = st;
    }
    { // stage V transposed: Bt[f][j] = V[g][mc+j][f]; pack adjacent j -> b32
      const float4* r0 = (const float4*)(V + (g * N_ + mc + vj0) * F_ + vf0);
      const float4* r1 = (const float4*)(V + (g * N_ + mc + vj0 + 1) * F_ + vf0);
      float4 x0 = r0[0], x1 = r0[1], x2 = r0[2], x3 = r0[3];
      float4 y0 = r1[0], y1 = r1[1], y2 = r1[2], y3 = r1[3];
      unsigned* dst = (unsigned*)(Bt + vf0 * 32 + vj0);
      dst[0]   = pack2bf(x0.x, y0.x); dst[16]  = pack2bf(x0.y, y0.y);
      dst[32]  = pack2bf(x0.z, y0.z); dst[48]  = pack2bf(x0.w, y0.w);
      dst[64]  = pack2bf(x1.x, y1.x); dst[80]  = pack2bf(x1.y, y1.y);
      dst[96]  = pack2bf(x1.z, y1.z); dst[112] = pack2bf(x1.w, y1.w);
      dst[128] = pack2bf(x2.x, y2.x); dst[144] = pack2bf(x2.y, y2.y);
      dst[160] = pack2bf(x2.z, y2.z); dst[176] = pack2bf(x2.w, y2.w);
      dst[192] = pack2bf(x3.x, y3.x); dst[208] = pack2bf(x3.y, y3.y);
      dst[224] = pack2bf(x3.z, y3.z); dst[240] = pack2bf(x3.w, y3.w);
      if (mc + 32 < N_) __builtin_prefetch((const float*)r0 + 32 * F_, 0, 3);
    }
    __syncthreads();
    v16bf af = load_frag_A16x32(At, lane);
    for (int i = 0; i < 2; ++i) {
      v16bf bfv = load_frag_B32x16(Bt, lane, (2 * w + i) * 16);
      acc[i] = __builtin_amdgcn_wmma_f32_16x16x32_bf16(false, af, false, bfv,
                                                       (short)0, acc[i], false, false);
    }
    __syncthreads();
  }

  for (int i = 0; i < 2; ++i) {
    int col = (2 * w + i) * 16 + (lane & 15);
    for (int rr = 0; rr < 8; ++rr) {
      int row = rr + ((lane >> 4) << 3);
      int n = n0 + row;
      float v = acc[i][rr];
      if (MODE == 0) {
        v = v > 0.f ? v : (__expf(v) - 1.f);   // ELU
        ((unsigned short*)Out)[(b * N_ + n) * (H_ * F_) + head * F_ + col] = f2bf(v);
      } else {
        ((float*)Out)[(g * N_ + n) * F_ + col] = v;
      }
    }
  }
}

// ---------------------------------------------------------------------------
// Residual add + LayerNorm (+ optional ReLU). One wave per 128-wide row.
// ---------------------------------------------------------------------------
__global__ void __launch_bounds__(256)
ln_kernel(const float* __restrict__ xin, const float* __restrict__ res,
          const float* __restrict__ gam, const float* __restrict__ bet,
          float* __restrict__ out, int relu, int rows)
{
  int t = threadIdx.x;
  int wid = (int)(blockIdx.x * blockDim.x + t) >> 5;
  if (wid >= rows) return;
  int lane = t & 31;
  float v4[4];
  float s = 0.f, ss = 0.f;
  for (int k = 0; k < 4; ++k) {
    int f = lane + 32 * k;
    float v = xin[wid * F_ + f] + res[wid * F_ + f];
    v4[k] = v; s += v; ss += v * v;
  }
  for (int d = 16; d > 0; d >>= 1) {
    s  += __shfl_xor(s, d, 32);
    ss += __shfl_xor(ss, d, 32);
  }
  float mean = s * (1.f / F_);
  float var  = ss * (1.f / F_) - mean * mean;
  float rstd = rsqrtf(var + 1e-5f);
  for (int k = 0; k < 4; ++k) {
    int f = lane + 32 * k;
    float y = (v4[k] - mean) * rstd * gam[f] + bet[f];
    if (relu) y = y > 0.f ? y : 0.f;
    out[wid * F_ + f] = y;
  }
}

extern "C" void kernel_launch(void* const* d_in, const int* in_sizes, int n_in,
                              void* d_out, int out_size, void* d_ws, size_t ws_size,
                              hipStream_t stream)
{
  (void)in_sizes; (void)n_in; (void)out_size; (void)ws_size;
  const float* x   = (const float*)d_in[0];
  const int*   adj = (const int*)  d_in[1];
  const float* Wp  = (const float*)d_in[2];
  const float* bp  = (const float*)d_in[3];
  const float* Wh  = (const float*)d_in[4];
  const float* ah  = (const float*)d_in[5];
  const float* Wo  = (const float*)d_in[6];
  const float* ao  = (const float*)d_in[7];
  const float* lg  = (const float*)d_in[8];
  const float* lb  = (const float*)d_in[9];

  char* ws = (char*)d_ws;
  float*          h     = (float*)(ws + 0);                      //  2 MB [B,N,F]
  float*          hH    = (float*)(ws + (2ull  << 20));          // 16 MB [B,H,N,F]
  unsigned short* multi = (unsigned short*)(ws + (18ull << 20)); //  8 MB bf16 [B,N,H*F]
  float*          hs    = (float*)(ws + (26ull << 20));          //  2 MB [B,N,F]
  float*          aoutb = (float*)(ws + (28ull << 20));          //  2 MB [B,N,F]
  float*          s1h   = (float*)(ws + (30ull << 20));
  float*          s2h   = s1h + B_ * H_ * N_;
  float*          s1s   = s2h + B_ * H_ * N_;
  float*          s2s   = s1s + B_ * N_;

  // input projection + bias + ReLU: h = relu(x @ Wp + bp)
  gemm128<0,1><<<dim3((B_ * N_) / 64, 1), 256, 0, stream>>>(
      x, Wp, h, bp, B_ * N_, DIN_, 1, 0, 0, 0);

  for (int l = 0; l < L_; ++l) {
    // per-head projections: hH[b,hd] = h[b] @ W_heads[l,hd]
    gemm128<0,0><<<dim3(N_ / 64, B_ * H_), 256, 0, stream>>>(
        h, Wh + l * H_ * F_ * F_, hH, nullptr,
        N_, F_, H_, N_ * F_, F_ * F_, N_ * F_);
    // head attention scores
    score_kernel<<<(B_ * H_ * N_) / 8, 256, 0, stream>>>(
        hH, ah + l * H_ * 2 * F_, s1h, s2h, H_, B_ * H_ * N_);
    // fused masked softmax * V, ELU, concat -> multi (bf16)
    attn_kernel<0><<<dim3(N_ / 16, B_ * H_), 128, 0, stream>>>(
        s1h, s2h, adj, hH, multi, H_);
    // out projection: hs = multi @ W_out[l]   (A is bf16, async-LDS staged)
    gemm128<1,0><<<dim3((B_ * N_) / 64, 1), 256, 0, stream>>>(
        multi, Wo + l * (H_ * F_) * F_, hs, nullptr,
        B_ * N_, H_ * F_, 1, 0, 0, 0);
    // single-head attention
    score_kernel<<<(B_ * N_) / 8, 256, 0, stream>>>(
        hs, ao + l * 2 * F_, s1s, s2s, 1, B_ * N_);
    attn_kernel<1><<<dim3(N_ / 16, B_), 128, 0, stream>>>(
        s1s, s2s, adj, hs, aoutb, 1);
    // residual + LayerNorm (+ReLU except last layer)
    float* dst = (l == L_ - 1) ? (float*)d_out : h;
    ln_kernel<<<(B_ * N_ * 32) / 256, 256, 0, stream>>>(
        aoutb, h, lg + l * F_, lb + l * F_, dst, (l < L_ - 1) ? 1 : 0, B_ * N_);
  }
}
